// Attention_50053548868012
// MI455X (gfx1250) — matrix operand order repaired and checked by reading, not compiled, not run on hardware
//
#include <hip/hip_runtime.h>

// ---------------------------------------------------------------------------
// GQA attention (B=1, S=2048, M=4096, 32 Q heads / 8 KV heads, H=128) + RoPE.
// All matmuls run on v_wmma_f32_16x16x32_bf16 (wave32, gfx1250).
// ---------------------------------------------------------------------------

typedef __attribute__((ext_vector_type(16))) __bf16 bf16x16;
typedef __attribute__((ext_vector_type(8)))  __bf16 bf16x8;
typedef __attribute__((ext_vector_type(8)))  float  f32x8;

#define S_LEN 2048
#define MDIM  4096
#define HDIM  128
#define NQH   32
#define NKV   8
#define QKVN  6144   // 4096 q + 1024 k + 1024 v columns

static __device__ __forceinline__ f32x8 wmma_bf16(bf16x16 a, bf16x16 b, f32x8 c) {
  // (neg_a, A, neg_b, B, c_mod, C, reuse_a, reuse_b)
  return __builtin_amdgcn_wmma_f32_16x16x32_bf16(false, a, false, b, (short)0, c,
                                                 false, false);
}

// 16-bit A 16x32 fragment from row-major A (lda in elements).
// lanes 0-15: row=lane, K {0..7,16..23}; lanes 16-31: row=lane-16, K {8..15,24..31}
static __device__ __forceinline__ bf16x16 load_a_frag(const __bf16* A, int lda) {
  const int lane = threadIdx.x & 31;
  const int m    = lane & 15;
  const int koff = (lane >> 4) * 8;
  const __bf16* p = A + (size_t)m * lda + koff;
  union { bf16x16 v; bf16x8 h[2]; } u;
  u.h[0] = *(const bf16x8*)(p);
  u.h[1] = *(const bf16x8*)(p + 16);
  return u.v;
}

// 16-bit B 32x16 fragment from B^T stored row-major [N][K] (ldb in elements).
// lanes 0-15: col n=lane, K {0..7,16..23}; lanes 16-31: col n=lane-16, K {8..15,24..31}
static __device__ __forceinline__ bf16x16 load_bt_frag(const __bf16* BT, int ldb) {
  const int lane = threadIdx.x & 31;
  const int n    = lane & 15;
  const int koff = (lane >> 4) * 8;
  const __bf16* p = BT + (size_t)n * ldb + koff;
  union { bf16x16 v; bf16x8 h[2]; } u;
  u.h[0] = *(const bf16x8*)(p);
  u.h[1] = *(const bf16x8*)(p + 16);
  return u.v;
}

// ---------------------------------------------------------------------------
// Stage 1 helpers: dtype conversion + weight packing (transposed bf16)
// ---------------------------------------------------------------------------
__global__ void cvt_f32_bf16(const float* __restrict__ src, __bf16* __restrict__ dst, int n) {
  int i = blockIdx.x * 256 + threadIdx.x;
  if (i < n) dst[i] = (__bf16)src[i];
}

// wqkvT[n][m] bf16, n in [0,6144): q cols (r*8+kv)*128+h, then k (1024), then v (1024)
__global__ void pack_wqkvT(const float* __restrict__ wq, const float* __restrict__ wk,
                           const float* __restrict__ wv, __bf16* __restrict__ wt) {
  int m = blockIdx.x * 256 + threadIdx.x;   // 0..4095
  int n = blockIdx.y;                       // 0..6143
  float v;
  if (n < 4096)       v = wq[(size_t)m * 4096 + n];
  else if (n < 5120)  v = wk[(size_t)m * 1024 + (n - 4096)];
  else                v = wv[(size_t)m * 1024 + (n - 5120)];
  wt[(size_t)n * MDIM + m] = (__bf16)v;
}

// woT[n][k] = wo_flat[k][n], wo is [R,KV,H,M] = [4096][4096] row-major
__global__ void pack_woT(const float* __restrict__ wo, __bf16* __restrict__ wt) {
  int k = blockIdx.x * 256 + threadIdx.x;   // 0..4095
  int n = blockIdx.y;                       // 0..4095
  wt[(size_t)n * MDIM + k] = (__bf16)wo[(size_t)k * MDIM + n];
}

// ---------------------------------------------------------------------------
// Generic WMMA GEMM: C[M x N] (f32) = A[M x K] (bf16, row-major) * B, with
// B supplied transposed: BT[N x K] (bf16, row-major).
// Block = 8 waves arranged 4(M) x 2(N); block tile 128 x 128.
// Each wave owns a 32 x 64 strip: 2 A-frags x 4 B-frags -> 8 WMMAs per K-step.
// All 6 fragment loads are issued into distinct registers before the WMMA
// burst so the scheduler can overlap load latency with earlier WMMAs.
// ---------------------------------------------------------------------------
__global__ __launch_bounds__(256) void gemm_bf16(const __bf16* __restrict__ A,
                                                 const __bf16* __restrict__ BT,
                                                 float* __restrict__ C,
                                                 int Mdim, int Ndim, int Kdim) {
  const int wave = threadIdx.x >> 5;
  const int lane = threadIdx.x & 31;
  const int wm   = wave & 3;               // 0..3 -> M strip
  const int wn   = wave >> 2;              // 0..1 -> N strip
  const int row0 = blockIdx.y * 128 + wm * 32;
  const int col0 = blockIdx.x * 128 + wn * 64;

  f32x8 acc[2][4] = {};
  for (int k = 0; k < Kdim; k += 32) {
    const __bf16* ap = A + (size_t)row0 * Kdim + k;
    __builtin_prefetch(ap + 64, 0, 1);                 // global_prefetch next K chunk
    bf16x16 a0 = load_a_frag(ap, Kdim);
    bf16x16 a1 = load_a_frag(ap + (size_t)16 * Kdim, Kdim);
    bf16x16 b[4];
#pragma unroll
    for (int j = 0; j < 4; ++j)
      b[j] = load_bt_frag(BT + (size_t)(col0 + j * 16) * Kdim + k, Kdim);
#pragma unroll
    for (int j = 0; j < 4; ++j) {
      acc[0][j] = wmma_bf16(a0, b[j], acc[0][j]);
      acc[1][j] = wmma_bf16(a1, b[j], acc[1][j]);
    }
  }
  // C layout: VGPR r holds rows {r, r+8}; lane%16 = column, lane/16 picks row half
  const int n = lane & 15, g = lane >> 4;
#pragma unroll
  for (int t = 0; t < 2; ++t)
#pragma unroll
    for (int j = 0; j < 4; ++j)
#pragma unroll
      for (int r = 0; r < 8; ++r)
        C[(size_t)(row0 + t * 16 + r + 8 * g) * Ndim + col0 + j * 16 + n] = acc[t][j][r];
}

// ---------------------------------------------------------------------------
// Stage 3: RoPE + split/layout. Reads qkv f32 [S][6144]; writes
//   qh [32][S][128] bf16 (roped), kh [8][S][128] bf16 (roped * H^-0.5),
//   vT [8][128][S] bf16 (transposed -> B^T operand for P*V).
// ---------------------------------------------------------------------------
__global__ void rope_split(const float* __restrict__ qkv,
                           const float* __restrict__ sinT, const float* __restrict__ cosT,
                           __bf16* __restrict__ qh, __bf16* __restrict__ kh,
                           __bf16* __restrict__ vT) {
  int c = blockIdx.x * 256 + threadIdx.x;   // 0..6143
  int s = blockIdx.y;                       // 0..2047
  float v = qkv[(size_t)s * QKVN + c];
  int h = c & 127;
  if (c < 5120) {
    float sn = sinT[(size_t)s * HDIM + h];
    float cs = cosT[(size_t)s * HDIM + h];
    float other = qkv[(size_t)s * QKVN + (h < 64 ? c + 64 : c - 64)];
    float rot = (h < 64) ? -other : other;
    float r = v * cs + rot * sn;
    if (c < 4096) {
      int hq = c >> 7;
      qh[((size_t)hq * S_LEN + s) * HDIM + h] = (__bf16)r;
    } else {
      int kv = (c - 4096) >> 7;
      kh[((size_t)kv * S_LEN + s) * HDIM + h] = (__bf16)(r * 0.08838834764831845f); // 128^-0.5
    }
  } else {
    int kv = (c - 5120) >> 7;
    vT[((size_t)kv * HDIM + h) * S_LEN + s] = (__bf16)v;
  }
}

// ---------------------------------------------------------------------------
// Stage 4: causal flash attention. One wave = one 16-query tile of one head.
// Block = 8 waves (128 queries); grid = (32 heads, 16 row blocks).
// Per 32-key step: 8 WMMAs for Q*K^T (K=128), LDS relayout of P (C-layout ->
// A-fragment), 8 WMMAs for P*V. Output written bf16 to yf[S][4096].
// ---------------------------------------------------------------------------
__global__ __launch_bounds__(256) void flash_attn(const __bf16* __restrict__ qh,
                                                  const __bf16* __restrict__ kh,
                                                  const __bf16* __restrict__ vT,
                                                  __bf16* __restrict__ yf) {
  __shared__ __bf16 smem[8][16][32];        // per-wave P staging (1 KB each)
  const int wave = threadIdx.x >> 5;
  const int lane = threadIdx.x & 31;
  const int hq   = blockIdx.x;              // 0..31, head = r*KV + kv
  const int kv   = hq & (NKV - 1);
  const int q0   = blockIdx.y * 128 + wave * 16;

  const __bf16* Q = qh + ((size_t)hq * S_LEN + q0) * HDIM;
  const __bf16* K = kh + (size_t)kv * S_LEN * HDIM;
  const __bf16* V = vT + (size_t)kv * HDIM * S_LEN;

  bf16x16 qf[4];
#pragma unroll
  for (int kc = 0; kc < 4; ++kc) qf[kc] = load_a_frag(Q + kc * 32, HDIM);

  f32x8 acc[8] = {};
  const float NEG = -__builtin_inff();
  float mst[8], lst[8];
#pragma unroll
  for (int r = 0; r < 8; ++r) { mst[r] = NEG; lst[r] = 0.f; }

  const int n = lane & 15, g = lane >> 4;

  for (int kb = 0; kb <= q0 + 15; kb += 32) {
    // ---- S = Q * K^T for 32 keys (two 16x16 C tiles) ----
    f32x8 s0 = {}, s1 = {};
#pragma unroll
    for (int kc = 0; kc < 4; ++kc) {
      bf16x16 b0 = load_bt_frag(K + (size_t)kb * HDIM + kc * 32, HDIM);
      bf16x16 b1 = load_bt_frag(K + (size_t)(kb + 16) * HDIM + kc * 32, HDIM);
      s0 = wmma_bf16(qf[kc], b0, s0);
      s1 = wmma_bf16(qf[kc], b1, s1);
    }
    // ---- causal mask + online softmax + stage P into LDS ----
#pragma unroll
    for (int r = 0; r < 8; ++r) {
      const int i = q0 + r + 8 * g;         // query index of this row
      float a0 = (kb + n      > i) ? NEG : s0[r];
      float a1 = (kb + 16 + n > i) ? NEG : s1[r];
      float mx = fmaxf(a0, a1);
#pragma unroll
      for (int d = 1; d < 16; d <<= 1) mx = fmaxf(mx, __shfl_xor(mx, d, 32));
      float mn    = fmaxf(mst[r], mx);
      float scale = __expf(mst[r] - mn);
      float e0 = __expf(a0 - mn);
      float e1 = __expf(a1 - mn);
      float sum = e0 + e1;
#pragma unroll
      for (int d = 1; d < 16; d <<= 1) sum += __shfl_xor(sum, d, 32);
      mst[r] = mn;
      lst[r] = lst[r] * scale + sum;
#pragma unroll
      for (int j = 0; j < 8; ++j) acc[j][r] *= scale;
      smem[wave][r + 8 * g][n]      = (__bf16)e0;
      smem[wave][r + 8 * g][16 + n] = (__bf16)e1;
    }
    // per-wave LDS RAW: DS ops are in-order per wave; stop compiler reordering
    asm volatile("s_wait_dscnt 0x0" ::: "memory");
    // ---- reload P in A-fragment layout ----
    union { bf16x16 v; bf16x8 h[2]; } pu;
    {
      const int m_ = lane & 15, ko = g * 8;
      pu.h[0] = *(const bf16x8*)&smem[wave][m_][ko];
      pu.h[1] = *(const bf16x8*)&smem[wave][m_][ko + 16];
    }
    // ---- acc += P * V (8 column tiles of H) ----
#pragma unroll
    for (int j = 0; j < 8; ++j) {
      bf16x16 vf = load_bt_frag(V + (size_t)(j * 16) * S_LEN + kb, S_LEN);
      acc[j] = wmma_bf16(pu.v, vf, acc[j]);
    }
  }

  // ---- normalize and store y[s][hq*128 + h] as bf16 ----
#pragma unroll
  for (int r = 0; r < 8; ++r) {
    float inv = 1.f / lst[r];
    const int srow = q0 + r + 8 * g;
#pragma unroll
    for (int j = 0; j < 8; ++j)
      yf[(size_t)srow * MDIM + hq * HDIM + j * 16 + n] = (__bf16)(acc[j][r] * inv);
  }
}

// ---------------------------------------------------------------------------
// Host-side orchestration. Inputs: x, wq, wk, wv, wo, mask(unused; computed
// on the fly), sin, cos — all f32. Output: f32 [2048][4096].
// ---------------------------------------------------------------------------
extern "C" void kernel_launch(void* const* d_in, const int* in_sizes, int n_in,
                              void* d_out, int out_size, void* d_ws, size_t ws_size,
                              hipStream_t stream) {
  (void)in_sizes; (void)n_in; (void)out_size; (void)ws_size;
  const float* x    = (const float*)d_in[0];
  const float* wq   = (const float*)d_in[1];
  const float* wk   = (const float*)d_in[2];
  const float* wv   = (const float*)d_in[3];
  const float* wo   = (const float*)d_in[4];
  const float* sinT = (const float*)d_in[6];
  const float* cosT = (const float*)d_in[7];
  float* out = (float*)d_out;

  char* ws = (char*)d_ws;                                   // offsets (MiB)
  __bf16* xb    = (__bf16*)(ws);                            //   0: 16 MiB
  __bf16* wqkvT = (__bf16*)(ws + (size_t)16  * (1 << 20));  //  16: 48 MiB
  __bf16* woT   = (__bf16*)(ws + (size_t)64  * (1 << 20));  //  64: 32 MiB
  float*  qkvf  = (float*) (ws + (size_t)96  * (1 << 20));  //  96: 48 MiB
  __bf16* yf    = (__bf16*)(ws + (size_t)96  * (1 << 20));  //  reuse qkvf (16 MiB)
  __bf16* qh    = (__bf16*)(ws + (size_t)144 * (1 << 20));  // 144: 16 MiB
  __bf16* kh    = (__bf16*)(ws + (size_t)160 * (1 << 20));  // 160:  4 MiB
  __bf16* vT    = (__bf16*)(ws + (size_t)164 * (1 << 20));  // 164:  4 MiB -> 168 total

  // 1) converts / packs
  cvt_f32_bf16<<<(S_LEN * MDIM) / 256, 256, 0, stream>>>(x, xb, S_LEN * MDIM);
  pack_wqkvT<<<dim3(MDIM / 256, QKVN), 256, 0, stream>>>(wq, wk, wv, wqkvT);
  pack_woT<<<dim3(MDIM / 256, MDIM), 256, 0, stream>>>(wo, woT);

  // 2) fused QKV projection: [2048 x 4096] * [4096 x 6144], 128x128 block tiles
  gemm_bf16<<<dim3(QKVN / 128, S_LEN / 128), 256, 0, stream>>>(
      xb, wqkvT, qkvf, S_LEN, QKVN, MDIM);

  // 3) RoPE + head split (+ K scale, V transpose)
  rope_split<<<dim3(QKVN / 256, S_LEN), 256, 0, stream>>>(qkvf, sinT, cosT, qh, kh, vT);

  // 4) causal GQA flash attention
  flash_attn<<<dim3(NQH, S_LEN / 128), 256, 0, stream>>>(qh, kh, vT, yf);

  // 5) output projection: [2048 x 4096] * [4096 x 4096], 128x128 block tiles
  gemm_bf16<<<dim3(MDIM / 128, S_LEN / 128), 256, 0, stream>>>(
      yf, woT, out, S_LEN, MDIM, MDIM);
}
